// GridCLIPModel_87136296501268
// MI455X (gfx1250) — matrix operand order repaired and checked by reading, not compiled, not run on hardware
//
#include <hip/hip_runtime.h>
#include <hip/hip_bf16.h>
#include <stdint.h>

// ---------------------------------------------------------------------------
// GridCLIP fused kernel for gfx1250 (MI455X, wave32, WMMA)
//   phase 0: hash-grid encode -> LDS f16 [64 x 128]; async-copy W1 -> LDS
//   phase 1: H1 = relu(G  @ W1 + b1)   (K=128, N=256)  -> LDS f16 [64 x 256]
//   phase 2: H2 = relu(H1 @ W2 + b2)   (K=256, N=256)  -> LDS f16 [64 x 256]
//   phase 3: OUT = H2 @ W3 + b3        (K=256, N=1024) -> global f32 (split)
// GEMMs on v_wmma_f32_16x16x32_f16; W1 staged via global_load_async_to_lds.
// ---------------------------------------------------------------------------

#define N_POINTS   262144
#define TILE_M     64          // points per workgroup
#define D_IN       128         // 16 levels * 8 dims
#define MLP_W      256
#define D_OUT      1024

typedef __attribute__((ext_vector_type(16))) _Float16 v16h;
typedef __attribute__((ext_vector_type(8)))  _Float16 v8h;
typedef __attribute__((ext_vector_type(8)))  float    v8f;

union Frag16 { v16h v; v8h h[2]; };

// hash-level layout: sizes are 2^min(21, 12+3l) -> power-of-two masks
__constant__ uint32_t LMASK[16] = {
    4096u - 1u, 32768u - 1u, 262144u - 1u,
    2097152u - 1u, 2097152u - 1u, 2097152u - 1u, 2097152u - 1u,
    2097152u - 1u, 2097152u - 1u, 2097152u - 1u, 2097152u - 1u,
    2097152u - 1u, 2097152u - 1u, 2097152u - 1u, 2097152u - 1u,
    2097152u - 1u
};
__constant__ uint32_t LOFF[16] = {
    0u, 4096u, 36864u, 299008u,
    2396160u, 4493312u, 6590464u, 8687616u,
    10784768u, 12881920u, 14979072u, 17076224u,
    19173376u, 21270528u, 23367680u, 25464832u
};

// ---- CDNA5 async global->LDS (ASYNCcnt-tracked), ISA 10 / 15.18.3 --------
__device__ __forceinline__ void async_load_b128_to_lds(uint32_t lds_addr,
                                                       const void* gptr) {
    asm volatile("global_load_async_to_lds_b128 %0, %1, off"
                 :
                 : "v"(lds_addr), "v"((uint64_t)(uintptr_t)gptr)
                 : "memory");
}
__device__ __forceinline__ void wait_asynccnt0() {
    asm volatile("s_wait_asynccnt 0" ::: "memory");
}

// ---- fragment loaders (layouts per CDNA5 ISA 7.12.2) ----------------------

// A: 16x32 f16 tile from row-major [M x ldk] halves.
__device__ __forceinline__ v16h load_a_frag(const _Float16* sA, int ldk,
                                            int m0, int k0, int lane) {
    int r  = lane & 15;
    int hs = lane >> 4;
    const _Float16* p = sA + (m0 + r) * ldk + k0 + hs * 8;
    Frag16 f;
    f.h[0] = *(const v8h*)(p);        // K = k0+8*hs   .. +7
    f.h[1] = *(const v8h*)(p + 16);   // K = k0+16+8*hs.. +7
    return f.v;
}

// B: 32x16 f16 tile from weights stored transposed [N x ldk] halves ([out][in]).
__device__ __forceinline__ v16h load_b_frag(const _Float16* wT,
                                            int ldk, int n0, int k0, int lane) {
    int c  = lane & 15;
    int hs = lane >> 4;
    const _Float16* p = wT + (size_t)(n0 + c) * ldk + k0 + hs * 8;
    Frag16 f;
    f.h[0] = *(const v8h*)(p);        // K = k0+8*hs   .. +7
    f.h[1] = *(const v8h*)(p + 16);   // K = k0+16+8*hs.. +7
    return f.v;
}

// One 16-wide output column tile across all 4 M-tiles, relu+bias, f16 to LDS.
// C/D layout: vgpr v, lane l -> (m = v + 8*(l>>4), n = l&15).
template<int NK>
__device__ __forceinline__ void layer_tile_to_lds(
        const _Float16* sA, int lda,
        const _Float16* wT,
        const float* __restrict__ bias,
        _Float16* sOut, int ldo,
        int n0, int lane) {
    v16h bf[NK];
#pragma unroll
    for (int k = 0; k < NK; ++k)
        bf[k] = load_b_frag(wT, NK * 32, n0, k * 32, lane);

    const int n  = n0 + (lane & 15);
    const int hs = lane >> 4;
    const float bv = bias[n];

#pragma unroll
    for (int mt = 0; mt < TILE_M / 16; ++mt) {
        v8f acc = {};
#pragma unroll
        for (int k = 0; k < NK; ++k) {
            v16h a = load_a_frag(sA, lda, mt * 16, k * 32, lane);
            acc = __builtin_amdgcn_wmma_f32_16x16x32_f16(
                    false, a, false, bf[k], (short)0, acc, false, false);
        }
#pragma unroll
        for (int v = 0; v < 8; ++v) {
            float xv = acc[v] + bv;
            xv = xv > 0.0f ? xv : 0.0f;
            sOut[(mt * 16 + v + 8 * hs) * ldo + n] = (_Float16)xv;
        }
    }
}

// Final layer: bias (no relu), f32 stores split into label/image halves.
template<int NK>
__device__ __forceinline__ void layer_tile_final(
        const _Float16* sA, int lda,
        const _Float16* __restrict__ wT,
        const float* __restrict__ bias,
        float* __restrict__ outLabel,   // [N_POINTS][512]
        float* __restrict__ outImage,   // [N_POINTS][512]
        int pointBase, int n0, int lane) {
    v16h bf[NK];
#pragma unroll
    for (int k = 0; k < NK; ++k)
        bf[k] = load_b_frag(wT, NK * 32, n0, k * 32, lane);

    const int n  = n0 + (lane & 15);
    const int hs = lane >> 4;
    const float bv = bias[n];
    float* dst = (n < 512) ? outLabel : outImage;
    const int col = (n < 512) ? n : (n - 512);

#pragma unroll
    for (int mt = 0; mt < TILE_M / 16; ++mt) {
        v8f acc = {};
#pragma unroll
        for (int k = 0; k < NK; ++k) {
            v16h a = load_a_frag(sA, lda, mt * 16, k * 32, lane);
            acc = __builtin_amdgcn_wmma_f32_16x16x32_f16(
                    false, a, false, bf[k], (short)0, acc, false, false);
        }
#pragma unroll
        for (int v = 0; v < 8; ++v) {
            int p = pointBase + mt * 16 + v + 8 * hs;
            dst[(size_t)p * 512 + col] = acc[v] + bv;
        }
    }
}

// ---------------------------------------------------------------------------
// weight repack: f32 row-major [in][out]  ->  f16 [out][in]
// ---------------------------------------------------------------------------
__global__ void pack_weights_kernel(const float* __restrict__ W1,
                                    const float* __restrict__ W2,
                                    const float* __restrict__ W3,
                                    _Float16* __restrict__ w1t,
                                    _Float16* __restrict__ w2t,
                                    _Float16* __restrict__ w3t) {
    int i = blockIdx.x * 256 + threadIdx.x;
    if (i < D_IN * MLP_W) {               // W1: [128][256]
        int k = i >> 8, n = i & 255;
        w1t[n * D_IN + k] = (_Float16)W1[i];
    }
    if (i < MLP_W * MLP_W) {              // W2: [256][256]
        int k = i >> 8, n = i & 255;
        w2t[n * MLP_W + k] = (_Float16)W2[i];
    }
    if (i < MLP_W * D_OUT) {              // W3: [256][1024]
        int k = i >> 10, n = i & 1023;
        w3t[(size_t)n * MLP_W + k] = (_Float16)W3[i];
    }
}

// ---------------------------------------------------------------------------
// fused encode + MLP
// ---------------------------------------------------------------------------
__global__ __launch_bounds__(256)
void gridclip_fused_kernel(const float* __restrict__ x,
                           const float* __restrict__ emb,
                           const _Float16* __restrict__ w1t,
                           const float* __restrict__ b1,
                           const _Float16* __restrict__ w2t,
                           const float* __restrict__ b2,
                           const _Float16* __restrict__ w3t,
                           const float* __restrict__ b3,
                           float* __restrict__ out) {
    __shared__ _Float16 sG [TILE_M * D_IN];    // 16 KB
    __shared__ _Float16 sH1[TILE_M * MLP_W];   // 32 KB
    __shared__ _Float16 sH2[TILE_M * MLP_W];   // 32 KB
    __shared__ _Float16 sW1[MLP_W * D_IN];     // 64 KB (W1 transposed [256][128])

    const int tid  = threadIdx.x;
    const int wid  = tid >> 5;
    const int lane = tid & 31;
    const int pointBase = blockIdx.x * TILE_M;

    // ---- kick off async W1 -> LDS copy; hides under hash-gather latency ---
    {
        const uint32_t ldsBase = (uint32_t)(uintptr_t)&sW1[0];
        const char* gsrc = (const char*)w1t;
#pragma unroll
        for (int it = 0; it < 16; ++it) {          // 16 * 256 lanes * 16B = 64KB
            uint32_t off = (uint32_t)(it * 4096 + tid * 16);
            async_load_b128_to_lds(ldsBase + off, gsrc + off);
        }
    }

    // ---------------- phase 0: hash-grid encode -----------------------
    // 4 threads per point, each handles 4 of the 16 levels.
    {
        const int p  = tid >> 2;       // local point 0..63
        const int lg = tid & 3;        // level group
        const size_t gp = (size_t)(pointBase + p);

        // bounded = (x+10)/20 in [0,1]; kernel remap u = (bounded+1)/2
        float ux = ((x[gp * 3 + 0] + 10.0f) * 0.05f + 1.0f) * 0.5f;
        float uy = ((x[gp * 3 + 1] + 10.0f) * 0.05f + 1.0f) * 0.5f;
        float uz = ((x[gp * 3 + 2] + 10.0f) * 0.05f + 1.0f) * 0.5f;

#pragma unroll
        for (int li = 0; li < 4; ++li) {
            const int l = lg * 4 + li;
            const uint32_t mask = LMASK[l];
            const uint32_t off  = LOFF[l];
            const float scale = (float)(16u << l) - 1.0f;

            float px = ux * scale + 0.5f;
            float py = uy * scale + 0.5f;
            float pz = uz * scale + 0.5f;
            float bx = floorf(px), by = floorf(py), bz = floorf(pz);
            float fx = px - bx, fy = py - by, fz = pz - bz;
            uint32_t ix = (uint32_t)bx, iy = (uint32_t)by, iz = (uint32_t)bz;

            float a0 = 0.f, a1 = 0.f, a2 = 0.f, a3 = 0.f;
            float a4 = 0.f, a5 = 0.f, a6 = 0.f, a7 = 0.f;
#pragma unroll
            for (int c = 0; c < 8; ++c) {
                uint32_t cx = ix + (c & 1);
                uint32_t cy = iy + ((c >> 1) & 1);
                uint32_t cz = iz + ((c >> 2) & 1);
                float w = ((c & 1)        ? fx : 1.0f - fx)
                        * (((c >> 1) & 1) ? fy : 1.0f - fy)
                        * (((c >> 2) & 1) ? fz : 1.0f - fz);
                uint32_t h = cx ^ (cy * 2654435761u) ^ (cz * 805459861u);
                size_t idx = (size_t)((h & mask) + off);
                const float4* e = reinterpret_cast<const float4*>(emb) + idx * 2;
                float4 e0 = e[0];
                float4 e1 = e[1];
                a0 += w * e0.x; a1 += w * e0.y; a2 += w * e0.z; a3 += w * e0.w;
                a4 += w * e1.x; a5 += w * e1.y; a6 += w * e1.z; a7 += w * e1.w;
            }
            _Float16* dst = &sG[p * D_IN + l * 8];
            dst[0] = (_Float16)a0; dst[1] = (_Float16)a1;
            dst[2] = (_Float16)a2; dst[3] = (_Float16)a3;
            dst[4] = (_Float16)a4; dst[5] = (_Float16)a5;
            dst[6] = (_Float16)a6; dst[7] = (_Float16)a7;
        }
    }
    wait_asynccnt0();     // W1 tile resident in LDS
    __syncthreads();

    // ---------------- phase 1: H1 = relu(G @ W1 + b1) ------------------
    // 16 column tiles over 8 waves -> 2 each; B fragments from LDS (sW1).
    layer_tile_to_lds<4>(sG, D_IN, sW1, b1, sH1, MLP_W, (wid * 2 + 0) * 16, lane);
    layer_tile_to_lds<4>(sG, D_IN, sW1, b1, sH1, MLP_W, (wid * 2 + 1) * 16, lane);
    __syncthreads();

    // ---------------- phase 2: H2 = relu(H1 @ W2 + b2) -----------------
    layer_tile_to_lds<8>(sH1, MLP_W, w2t, b2, sH2, MLP_W, (wid * 2 + 0) * 16, lane);
    layer_tile_to_lds<8>(sH1, MLP_W, w2t, b2, sH2, MLP_W, (wid * 2 + 1) * 16, lane);
    __syncthreads();

    // ---------------- phase 3: OUT = H2 @ W3 + b3 ----------------------
    // 64 column tiles over 8 waves -> 8 each; split label/image halves.
    float* outLabel = out;
    float* outImage = out + (size_t)N_POINTS * 512;
#pragma unroll
    for (int i = 0; i < 8; ++i) {
        layer_tile_final<8>(sH2, MLP_W, w3t, b3, outLabel, outImage,
                            pointBase, (wid * 8 + i) * 16, lane);
    }
}

// ---------------------------------------------------------------------------
extern "C" void kernel_launch(void* const* d_in, const int* in_sizes, int n_in,
                              void* d_out, int out_size, void* d_ws, size_t ws_size,
                              hipStream_t stream) {
    const float* x   = (const float*)d_in[0];
    const float* emb = (const float*)d_in[1];
    const float* W1  = (const float*)d_in[2];
    const float* b1  = (const float*)d_in[3];
    const float* W2  = (const float*)d_in[4];
    const float* b2  = (const float*)d_in[5];
    const float* W3  = (const float*)d_in[6];
    const float* b3  = (const float*)d_in[7];
    float* out = (float*)d_out;

    _Float16* w1t = (_Float16*)d_ws;            //  64 KB
    _Float16* w2t = w1t + D_IN * MLP_W;         // 128 KB
    _Float16* w3t = w2t + MLP_W * MLP_W;        // 512 KB

    // repack weights to f16 [out][in] (covers the largest range: 256*1024)
    pack_weights_kernel<<<(MLP_W * D_OUT) / 256, 256, 0, stream>>>(
        W1, W2, W3, w1t, w2t, w3t);

    gridclip_fused_kernel<<<N_POINTS / TILE_M, 256, 0, stream>>>(
        x, emb, w1t, b1, w2t, b2, w3t, b3, out);
}
